// GATDDI_43087111914332
// MI455X (gfx1250) — compile-verified, matrix operand order checked by reading, not run on hardware
//
#include <hip/hip_runtime.h>
#include <math.h>

typedef __attribute__((ext_vector_type(16))) _Float16 v16h;
typedef __attribute__((ext_vector_type(8)))  _Float16 v8h;
typedef __attribute__((ext_vector_type(8)))  float    v8f;

// ---------------- problem constants (from reference) ----------------
#define NN    20000
#define EE    640000
#define PP    200000
#define IND   256
#define HIDC  64
#define HDS   4
#define OUTD  86
#define OUTP  96             // OUTD padded to multiple of 16
#define ETOT  (EE + NN)      // edges + self loops
#define HC1   (HDS * HIDC)   // 256

// ---------------- helpers ----------------
__device__ __forceinline__ unsigned f2ord(float f) {
  unsigned u = __float_as_uint(f);
  return (u & 0x80000000u) ? ~u : (u | 0x80000000u);
}
__device__ __forceinline__ float ord2f(unsigned u) {
  unsigned b = (u & 0x80000000u) ? (u & 0x7FFFFFFFu) : ~u;
  return __uint_as_float(b);
}

// A-fragment (ISA 7.12.2, 16-bit A 16x32): lane m+16*half holds row m,
// K = kb + i (i<8) and K = kb + 8 + i (i>=8), kb = kk*32 + 8*half.
// p = row base + kb, f16 row-major, 16B aligned.
__device__ __forceinline__ v16h load_a_f16(const _Float16* p) {
  v8h lo = *reinterpret_cast<const v8h*>(p);
  v8h hi = *reinterpret_cast<const v8h*>(p + 16);
  return __builtin_shufflevector(lo, hi, 0,1,2,3,4,5,6,7,8,9,10,11,12,13,14,15);
}

// ---------------- prep: pack B into per-lane fragment layout ----------------
// Bp[((nt*(K/32)+kk)*32+lane)*16 + i] = W[kg*Nreal + col] (0 if col>=Nreal)
// kg = kk*32 + 8*(lane>>4) + i + (i>=8?8:0), col = nt*16 + (lane&15)
__global__ void pack_b_kernel(const float* __restrict__ W, _Float16* __restrict__ Bp,
                              int K, int Npad, int Nreal) {
  int gid = blockIdx.x * blockDim.x + threadIdx.x;
  if (gid >= Npad * K) return;
  int i    = gid & 15;
  int lane = (gid >> 4) & 31;
  int frag = gid >> 9;           // nt*(K/32)+kk
  int KT   = K >> 5;
  int kk   = frag % KT;
  int nt   = frag / KT;
  int kg   = kk * 32 + 8 * (lane >> 4) + i + ((i >= 8) ? 8 : 0);
  int col  = nt * 16 + (lane & 15);
  Bp[gid] = (col < Nreal) ? (_Float16)W[(size_t)kg * Nreal + col] : (_Float16)0.0f;
}

// ---------------- prep: fp32 -> f16 row-major activation copy ---------------
__global__ void cvt_f16_kernel(const float* __restrict__ in, _Float16* __restrict__ out, int n) {
  int gid = blockIdx.x * blockDim.x + threadIdx.x;
  if (gid < n) out[gid] = (_Float16)in[gid];
}

// ---------------- generic WMMA GEMM: C[M,N] = A[M,K] * B[K,N] ----------------
// A: f16 row-major, Bp: packed fragments. grid (M/16, N/64), 1 wave/block.
__global__ void wmma_gemm_f16(const _Float16* __restrict__ A,
                              const _Float16* __restrict__ Bp,
                              float* __restrict__ C, int K, int N) {
  int lane = threadIdx.x;
  int half = lane >> 4;
  int mn   = lane & 15;
  int KT   = K >> 5;
  const _Float16* arow = A + (size_t)(blockIdx.x * 16 + mn) * K;
  const v16h* bp = reinterpret_cast<const v16h*>(Bp);

  v8f c[4] = {};
  for (int kk = 0; kk < KT; ++kk) {
    int kb = kk * 32 + half * 8;
    v16h a = load_a_f16(arow + kb);
#pragma unroll
    for (int nt = 0; nt < 4; ++nt) {
      int ntg = blockIdx.y * 4 + nt;
      v16h b = bp[(ntg * KT + kk) * 32 + lane];
      c[nt] = __builtin_amdgcn_wmma_f32_16x16x32_f16(false, a, false, b,
                                                     (short)0, c[nt], false, false);
    }
  }
#pragma unroll
  for (int nt = 0; nt < 4; ++nt) {
    int col = blockIdx.y * 64 + nt * 16 + mn;
#pragma unroll
    for (int r = 0; r < 8; ++r) {
      int row = blockIdx.x * 16 + half * 8 + r;
      C[(size_t)row * N + col] = c[nt][r];
    }
  }
}

// ---------------- alpha_s / alpha_d : per (node, head) dot products ----------
__global__ void alpha_kernel(const float* __restrict__ h,
                             const float* __restrict__ a_src,
                             const float* __restrict__ a_dst,
                             float* __restrict__ as_, float* __restrict__ ad_,
                             int n, int heads, int C) {
  int gid = blockIdx.x * blockDim.x + threadIdx.x;
  if (gid >= n * heads) return;
  int node = gid / heads, hd = gid - node * heads;
  const float* hp = h + (size_t)node * heads * C + hd * C;
  const float* s = a_src + hd * C;
  const float* d = a_dst + hd * C;
  float ss = 0.f, dd = 0.f;
  for (int c = 0; c < C; ++c) { float v = hp[c]; ss += v * s[c]; dd += v * d[c]; }
  as_[gid] = ss;
  ad_[gid] = dd;
}

__device__ __forceinline__ void edge_sd(const int* ei, int e, int& s, int& d) {
  if (e < EE) { s = ei[e]; d = ei[EE + e]; }
  else        { s = e - EE; d = e - EE; }   // self loops
}

// ---------------- pass 1: segment max (order-preserving u32 atomicMax) -------
__global__ void edge_max_kernel(const int* __restrict__ ei,
                                const float* __restrict__ as_,
                                const float* __restrict__ ad_,
                                unsigned* __restrict__ mbuf, int heads) {
  int gid = blockIdx.x * blockDim.x + threadIdx.x;
  if (gid >= ETOT * heads) return;
  int e = gid / heads, hd = gid - e * heads;
  int s, d; edge_sd(ei, e, s, d);
  float v = as_[s * heads + hd] + ad_[d * heads + hd];
  v = (v > 0.f) ? v : 0.2f * v;   // leaky_relu(0.2)
  atomicMax(&mbuf[d * heads + hd], f2ord(v));
}

// ---------------- pass 2: exp(e - max) and segment sum -----------------------
__global__ void edge_exp_kernel(const int* __restrict__ ei,
                                const float* __restrict__ as_,
                                const float* __restrict__ ad_,
                                const unsigned* __restrict__ mbuf,
                                float* __restrict__ expv,
                                float* __restrict__ denom, int heads) {
  int gid = blockIdx.x * blockDim.x + threadIdx.x;
  if (gid >= ETOT * heads) return;
  int e = gid / heads, hd = gid - e * heads;
  int s, d; edge_sd(ei, e, s, d);
  float v = as_[s * heads + hd] + ad_[d * heads + hd];
  v = (v > 0.f) ? v : 0.2f * v;
  float ex = __expf(v - ord2f(mbuf[d * heads + hd]));
  expv[gid] = ex;
  atomicAdd(&denom[d * heads + hd], ex);
}

// ---------------- pass 3: weighted message scatter-add -----------------------
__global__ void edge_msg_kernel(const int* __restrict__ ei,
                                const float* __restrict__ h,
                                const float* __restrict__ expv,
                                const float* __restrict__ denom,
                                float* __restrict__ acc, int heads, int C, int HC) {
  int segs = HC >> 4;
  int gid = blockIdx.x * blockDim.x + threadIdx.x;
  if (gid >= ETOT * segs) return;
  int e = gid / segs, sg = gid - e * segs;
  int base = sg * 16;
  int hd = base / C;
  int s, d; edge_sd(ei, e, s, d);
  float w = expv[e * heads + hd] / denom[d * heads + hd];
  const float* hs = h + (size_t)s * HC + base;
  float* ap = acc + (size_t)d * HC + base;
#pragma unroll
  for (int i = 0; i < 16; ++i) atomicAdd(&ap[i], hs[i] * w);
}

// ---------------- finalize: acc = elu(acc + b) -------------------------------
__global__ void finalize_kernel(float* __restrict__ acc, const float* __restrict__ b,
                                int n, int dim) {
  int gid = blockIdx.x * blockDim.x + threadIdx.x;
  if (gid >= n * dim) return;
  int c = gid % dim;
  float x = acc[gid] + b[c];
  acc[gid] = (x > 0.f) ? x : (__expf(x) - 1.f);
}

// ---------------- fused pair MLP via two WMMA GEMMs --------------------------
// 1 wave / 16 pairs: z = relu([h[p0]||h[p1]] @ Wp1 + bp1);  out = z @ Wp2 + bp2
// hf: f16 node embeddings [NN,64]; Wp1p packed K=128,N=64; Wp2p packed K=64,N=96.
__global__ void pair_mlp_kernel(const _Float16* __restrict__ hf,
                                const int* __restrict__ pairs,
                                const _Float16* __restrict__ Wp1p,
                                const float* __restrict__ bp1,
                                const _Float16* __restrict__ Wp2p,
                                const float* __restrict__ bp2,
                                float* __restrict__ out) {
  __shared__ _Float16 zh[16][64];
  int lane = threadIdx.x;
  int half = lane >> 4;
  int mn   = lane & 15;
  int tile = blockIdx.x;
  int prow = tile * 16 + mn;
  int p0 = pairs[prow * 2];
  int p1 = pairs[prow * 2 + 1];
  const v16h* b1p = reinterpret_cast<const v16h*>(Wp1p);
  const v16h* b2p = reinterpret_cast<const v16h*>(Wp2p);

  // ---- GEMM1: K = 128 (p0 -> k 0..63, p1 -> k 64..127), N = 64 ----
  v8f c[4] = {};
#pragma unroll
  for (int kk = 0; kk < 4; ++kk) {
    const _Float16* row = (kk < 2) ? (hf + (size_t)p0 * HIDC) : (hf + (size_t)p1 * HIDC);
    v16h a = load_a_f16(row + (kk & 1) * 32 + half * 8);
#pragma unroll
    for (int nt = 0; nt < 4; ++nt) {
      v16h b = b1p[(nt * 4 + kk) * 32 + lane];
      c[nt] = __builtin_amdgcn_wmma_f32_16x16x32_f16(false, a, false, b,
                                                     (short)0, c[nt], false, false);
    }
  }
  // bias + relu, D-layout -> LDS (row-major f16, A-readable)
#pragma unroll
  for (int nt = 0; nt < 4; ++nt) {
    int col = nt * 16 + mn;
    float bb = bp1[col];
#pragma unroll
    for (int r = 0; r < 8; ++r) {
      float v = c[nt][r] + bb;
      v = (v > 0.f) ? v : 0.f;
      zh[r + 8 * half][col] = (_Float16)v;
    }
  }
  __syncthreads();

  // ---- GEMM2: K = 64, N = 86 padded to 96 ----
  v16h a2[2];
#pragma unroll
  for (int kk = 0; kk < 2; ++kk)
    a2[kk] = load_a_f16(&zh[mn][0] + kk * 32 + half * 8);

#pragma unroll
  for (int nt = 0; nt < 6; ++nt) {
    int col = nt * 16 + mn;
    v8f c2 = {};
#pragma unroll
    for (int kk = 0; kk < 2; ++kk) {
      v16h b = b2p[(nt * 2 + kk) * 32 + lane];
      c2 = __builtin_amdgcn_wmma_f32_16x16x32_f16(false, a2[kk], false, b,
                                                  (short)0, c2, false, false);
    }
    if (col < OUTD) {
      float bb = bp2[col];
#pragma unroll
      for (int r = 0; r < 8; ++r) {
        int row = tile * 16 + r + 8 * half;
        out[(size_t)row * OUTD + col] = c2[r] + bb;
      }
    }
  }
}

// ---------------- host orchestration ----------------------------------------
extern "C" void kernel_launch(void* const* d_in, const int* in_sizes, int n_in,
                              void* d_out, int out_size, void* d_ws, size_t ws_size,
                              hipStream_t stream) {
  (void)in_sizes; (void)n_in; (void)out_size; (void)ws_size;
  const float* x      = (const float*)d_in[0];
  const int*   ei     = (const int*)  d_in[1];
  const int*   pairs  = (const int*)  d_in[2];
  const float* W1     = (const float*)d_in[3];
  const float* a_src1 = (const float*)d_in[4];
  const float* a_dst1 = (const float*)d_in[5];
  const float* b1     = (const float*)d_in[6];
  const float* W2     = (const float*)d_in[7];
  const float* a_src2 = (const float*)d_in[8];
  const float* a_dst2 = (const float*)d_in[9];
  const float* b2     = (const float*)d_in[10];
  const float* Wp1    = (const float*)d_in[11];
  const float* bp1    = (const float*)d_in[12];
  const float* Wp2    = (const float*)d_in[13];
  const float* bp2    = (const float*)d_in[14];
  float* outp = (float*)d_out;

  // workspace layout (256B-aligned regions)
  char* base = (char*)d_ws;
  size_t o = 0;
  auto take = [&](size_t bytes) { char* p = base + o; o = (o + bytes + 255) & ~(size_t)255; return p; };
  float*     h1    = (float*)    take((size_t)NN * HC1 * 4);   // aliased as h2 later
  float*     acc1  = (float*)    take((size_t)NN * HC1 * 4);
  float*     expv  = (float*)    take((size_t)ETOT * HDS * 4); // reused for layer 2
  float*     as_   = (float*)    take((size_t)NN * HDS * 4);
  float*     ad_   = (float*)    take((size_t)NN * HDS * 4);
  unsigned*  mbuf  = (unsigned*) take((size_t)NN * HDS * 4);
  float*     denom = (float*)    take((size_t)NN * HDS * 4);
  float*     acc2  = (float*)    take((size_t)NN * HIDC * 4);
  _Float16*  af16  = (_Float16*) take((size_t)NN * HC1 * 2);   // f16 activations (x, then out1)
  _Float16*  hf    = (_Float16*) take((size_t)NN * HIDC * 2);  // f16 final node emb
  _Float16*  W1p   = (_Float16*) take((size_t)IND * HC1 * 2);
  _Float16*  W2p   = (_Float16*) take((size_t)HC1 * HIDC * 2);
  _Float16*  Wp1p  = (_Float16*) take((size_t)(2 * HIDC) * HIDC * 2);
  _Float16*  Wp2p  = (_Float16*) take((size_t)HIDC * OUTP * 2);
  float*     h2    = h1;

  const int T = 256;
  auto blks = [](long long n, int t) { return (unsigned)((n + t - 1) / t); };

  // ===== weight packing (once per launch) =====
  pack_b_kernel<<<blks((long long)HC1 * IND, T), T, 0, stream>>>(W1, W1p, IND, HC1, HC1);
  pack_b_kernel<<<blks((long long)HIDC * HC1, T), T, 0, stream>>>(W2, W2p, HC1, HIDC, HIDC);
  pack_b_kernel<<<blks((long long)HIDC * 2 * HIDC, T), T, 0, stream>>>(Wp1, Wp1p, 2 * HIDC, HIDC, HIDC);
  pack_b_kernel<<<blks((long long)OUTP * HIDC, T), T, 0, stream>>>(Wp2, Wp2p, HIDC, OUTP, OUTD);

  // ===== Layer 1 (heads=4, C=64, HC=256) =====
  hipMemsetAsync(acc1, 0, (size_t)NN * HC1 * 4, stream);
  hipMemsetAsync(mbuf, 0, (size_t)NN * HDS * 4, stream);
  hipMemsetAsync(denom, 0, (size_t)NN * HDS * 4, stream);

  cvt_f16_kernel<<<blks((long long)NN * IND, T), T, 0, stream>>>(x, af16, NN * IND);
  wmma_gemm_f16<<<dim3(NN / 16, HC1 / 64), 32, 0, stream>>>(af16, W1p, h1, IND, HC1);
  alpha_kernel<<<blks((long long)NN * HDS, T), T, 0, stream>>>(h1, a_src1, a_dst1, as_, ad_, NN, HDS, HIDC);
  edge_max_kernel<<<blks((long long)ETOT * HDS, T), T, 0, stream>>>(ei, as_, ad_, mbuf, HDS);
  edge_exp_kernel<<<blks((long long)ETOT * HDS, T), T, 0, stream>>>(ei, as_, ad_, mbuf, expv, denom, HDS);
  edge_msg_kernel<<<blks((long long)ETOT * (HC1 / 16), T), T, 0, stream>>>(ei, h1, expv, denom, acc1, HDS, HIDC, HC1);
  finalize_kernel<<<blks((long long)NN * HC1, T), T, 0, stream>>>(acc1, b1, NN, HC1);  // acc1 = out1

  // ===== Layer 2 (heads=1, C=64, HC=64) =====
  cvt_f16_kernel<<<blks((long long)NN * HC1, T), T, 0, stream>>>(acc1, af16, NN * HC1);
  wmma_gemm_f16<<<dim3(NN / 16, HIDC / 64), 32, 0, stream>>>(af16, W2p, h2, HC1, HIDC);
  hipMemsetAsync(mbuf, 0, (size_t)NN * 4, stream);
  hipMemsetAsync(denom, 0, (size_t)NN * 4, stream);
  hipMemsetAsync(acc2, 0, (size_t)NN * HIDC * 4, stream);
  alpha_kernel<<<blks((long long)NN, T), T, 0, stream>>>(h2, a_src2, a_dst2, as_, ad_, NN, 1, HIDC);
  edge_max_kernel<<<blks((long long)ETOT, T), T, 0, stream>>>(ei, as_, ad_, mbuf, 1);
  edge_exp_kernel<<<blks((long long)ETOT, T), T, 0, stream>>>(ei, as_, ad_, mbuf, expv, denom, 1);
  edge_msg_kernel<<<blks((long long)ETOT * (HIDC / 16), T), T, 0, stream>>>(ei, h2, expv, denom, acc2, 1, HIDC, HIDC);
  finalize_kernel<<<blks((long long)NN * HIDC, T), T, 0, stream>>>(acc2, b2, NN, HIDC); // acc2 = final emb

  // ===== Pair MLP (fused two WMMA GEMMs) =====
  cvt_f16_kernel<<<blks((long long)NN * HIDC, T), T, 0, stream>>>(acc2, hf, NN * HIDC);
  pair_mlp_kernel<<<PP / 16, 32, 0, stream>>>(hf, pairs, Wp1p, bp1, Wp2p, bp2, outp);
}
